// MoE_layer_85023172591904
// MI455X (gfx1250) — matrix operand order repaired and checked by reading, not compiled, not run on hardware
//
#include <hip/hip_runtime.h>
#include <math.h>

#define N_TOK 4096
#define DIM   1024
#define NEXP  8
#define CAP   8448       // max padded assignments: 8192 + 8*31 = 8440, rounded up
#define LDS_STRIDE 1032  // 1024 + 8 bf16 pad: 2064B row stride -> bank shift 4/row, 16B aligned

typedef unsigned short u16t;
typedef unsigned int   u32t;
typedef __attribute__((ext_vector_type(16))) __bf16 v16bf;
typedef __attribute__((ext_vector_type(16))) u16t   v16u;
typedef __attribute__((ext_vector_type(8)))  u16t   v8u;
typedef __attribute__((ext_vector_type(8)))  float  v8f;

union BF16V { v16u u; v16bf bf; };

__device__ __forceinline__ u16t f2bf(float f) {
  union { float f; u32t u; } c; c.f = f;
  u32t u = c.u;
  u += 0x7FFFu + ((u >> 16) & 1u);   // round-to-nearest-even
  return (u16t)(u >> 16);
}

__device__ __forceinline__ v16u pack16(v8u lo, v8u hi) {
  v16u r;
  #pragma unroll
  for (int i = 0; i < 8; ++i) { r[i] = lo[i]; r[8 + i] = hi[i]; }
  return r;
}

// Shared inner K-loop: 32x32 wave tile, A from LDS, B double-buffered from global.
// Declares/updates c00,c01,c10,c11 (v8f) in the enclosing scope.
#define GEMM_KLOOP(ATILE, WBF)                                                              \
  v8f c00 = {}, c01 = {}, c10 = {}, c11 = {};                                               \
  {                                                                                         \
    int koff = (lane & 16) ? 8 : 0;                                                         \
    int row0 = (lane & 15);                                                                 \
    const int nb0 = nBase >> 4;                                                             \
    const u16t* wb0 = (WBF) + ((size_t)(e * 32 * 64) + nb0) * 512 + lane * 16;              \
    v16u bcur0 = *(const v16u*)(wb0);                                                       \
    v16u bcur1 = *(const v16u*)(wb0 + 512);                                                 \
    _Pragma("unroll 2")                                                                     \
    for (int kb = 0; kb < 32; ++kb) {                                                       \
      int kn = (kb < 31) ? (kb + 1) : 31;                                                   \
      const u16t* wbn = (WBF) + ((size_t)((e * 32 + kn) * 64) + nb0) * 512 + lane * 16;     \
      v16u bn0 = *(const v16u*)(wbn);                                                       \
      v16u bn1 = *(const v16u*)(wbn + 512);                                                 \
      int k0 = kb * 32 + koff;                                                              \
      const u16t* ar0 = (ATILE) + row0 * LDS_STRIDE + k0;                                   \
      const u16t* ar1 = (ATILE) + (16 + row0) * LDS_STRIDE + k0;                            \
      BF16V a0; a0.u = pack16(*(const v8u*)(ar0), *(const v8u*)(ar0 + 16));                 \
      BF16V a1; a1.u = pack16(*(const v8u*)(ar1), *(const v8u*)(ar1 + 16));                 \
      BF16V b0; b0.u = bcur0;                                                               \
      BF16V b1; b1.u = bcur1;                                                               \
      c00 = __builtin_amdgcn_wmma_f32_16x16x32_bf16(false, a0.bf, false, b0.bf, (short)0, c00, false, false); \
      c10 = __builtin_amdgcn_wmma_f32_16x16x32_bf16(false, a1.bf, false, b0.bf, (short)0, c10, false, false); \
      c01 = __builtin_amdgcn_wmma_f32_16x16x32_bf16(false, a0.bf, false, b1.bf, (short)0, c01, false, false); \
      c11 = __builtin_amdgcn_wmma_f32_16x16x32_bf16(false, a1.bf, false, b1.bf, (short)0, c11, false, false); \
      bcur0 = bn0;                                                                          \
      bcur1 = bn1;                                                                          \
    }                                                                                       \
  }

// ---------------- init: zero meta + assignment arrays ----------------
__global__ void init_kernel(int* assignTok, float* assignGate, int* meta) {
  int i = blockIdx.x * blockDim.x + threadIdx.x;
  if (i < CAP) { assignTok[i] = 0; assignGate[i] = 0.0f; }
  if (i < 32)  meta[i] = 0;
}

// ---------------- x -> bf16 ----------------
__global__ void conv_x_kernel(const float* __restrict__ x, u16t* __restrict__ xbf) {
  int i = blockIdx.x * blockDim.x + threadIdx.x;   // N_TOK*DIM threads
  xbf[i] = f2bf(x[i]);
}

// ---------------- W[e,k,n] -> WMMA-B-fragment-native tiled bf16 ----------------
// tile = ((e*32 + kb)*64 + nb), 512 elems/tile.
// within tile: lane*16 + t, where col = lane&15, kst = (lane&16)?8:0, k = kb*32+kst+(t&7)+((t&8)<<1)
__global__ void prep_w_kernel(const float* __restrict__ W, u16t* __restrict__ Wbf) {
  int i = blockIdx.x * blockDim.x + threadIdx.x;   // 8*1024*1024 threads
  int r    = i & 511;
  int tile = i >> 9;
  int nb = tile & 63;
  int kb = (tile >> 6) & 31;
  int e  = tile >> 11;
  int lane = r >> 4;
  int t    = r & 15;
  int col = lane & 15;
  int kst = (lane & 16) ? 8 : 0;
  int k = kb * 32 + kst + (t & 7) + ((t & 8) << 1);
  Wbf[i] = f2bf(W[((size_t)e * DIM + k) * DIM + nb * 16 + col]);
}

// ---------------- gating: wave per token ----------------
__global__ void gate_kernel(const float* __restrict__ x, const float* __restrict__ wg,
                            int* __restrict__ topIdx, float* __restrict__ topGate,
                            int* __restrict__ cnt) {
  int wave = threadIdx.x >> 5;
  int lane = threadIdx.x & 31;
  int n = blockIdx.x * 8 + wave;
  float acc[NEXP];
  #pragma unroll
  for (int e = 0; e < NEXP; ++e) acc[e] = 0.0f;
  for (int k = 0; k < 32; ++k) {
    int d = k * 32 + lane;
    float xv = x[(size_t)n * DIM + d];
    const float* wrow = wg + d * NEXP;
    #pragma unroll
    for (int e = 0; e < NEXP; ++e) acc[e] += xv * wrow[e];
  }
  #pragma unroll
  for (int off = 16; off > 0; off >>= 1) {
    #pragma unroll
    for (int e = 0; e < NEXP; ++e) acc[e] += __shfl_xor(acc[e], off, 32);
  }
  if (lane == 0) {
    int i0 = 0;
    #pragma unroll
    for (int e = 1; e < NEXP; ++e) if (acc[e] > acc[i0]) i0 = e;
    int i1 = (i0 == 0) ? 1 : 0;
    #pragma unroll
    for (int e = 0; e < NEXP; ++e) if (e != i0 && acc[e] > acc[i1]) i1 = e;
    float e1 = expf(acc[i1] - acc[i0]);
    float inv = 1.0f / (1.0f + e1);
    topIdx[n * 2] = i0;  topIdx[n * 2 + 1] = i1;
    topGate[n * 2] = inv; topGate[n * 2 + 1] = e1 * inv;
    atomicAdd(&cnt[i0], 1);
    atomicAdd(&cnt[i1], 1);
  }
}

// ---------------- offsets: pad each expert to multiple of 32 ----------------
__global__ void offsets_kernel(int* meta) {
  if (threadIdx.x == 0) {
    int* cnt = meta; int* cntPad = meta + 8; int* offs = meta + 16;
    int tot = 0;
    for (int e = 0; e < NEXP; ++e) {
      offs[e] = tot;
      int pc = ((cnt[e] + 31) / 32) * 32;
      cntPad[e] = pc;
      tot += pc;
    }
  }
}

// ---------------- scatter tokens into expert segments ----------------
__global__ void scatter_kernel(const int* __restrict__ topIdx, const float* __restrict__ topGate,
                               int* meta, int* __restrict__ assignTok,
                               float* __restrict__ assignGate, int* __restrict__ slotPos) {
  int n = blockIdx.x * blockDim.x + threadIdx.x;   // N_TOK threads
  const int* offs = meta + 16; int* cursor = meta + 24;
  #pragma unroll
  for (int s = 0; s < 2; ++s) {
    int e = topIdx[n * 2 + s];
    int p = offs[e] + atomicAdd(&cursor[e], 1);
    assignTok[p] = n;
    assignGate[p] = topGate[n * 2 + s];
    slotPos[n * 2 + s] = p;
  }
}

// ---------------- GEMM1: H = gelu(Xg @ W1 + b1), bf16 out ----------------
// grid (4, 128, 8); block 256 = 8 waves; block tile M=32 x N=256; wave tile 32x32
__global__ void __launch_bounds__(256)
gemm1_kernel(const u16t* __restrict__ xbf, const u16t* __restrict__ w1bf,
             const float* __restrict__ bias1, const int* __restrict__ assignTok,
             const int* __restrict__ meta, u16t* __restrict__ hbuf) {
  __shared__ u16t atile[32 * LDS_STRIDE];
  int e = blockIdx.z;
  int pc = meta[8 + e];                      // cntPad
  int mt = blockIdx.y * 32;
  if (mt >= pc) return;
  int mBase = meta[16 + e] + mt;             // offs
  int wave = threadIdx.x >> 5;
  int lane = threadIdx.x & 31;
  int nBase = blockIdx.x * 256 + wave * 32;

  // cooperative gathered fill of A tile: 32 rows x 1024 cols, 8-bf16 chunks
  for (int it = threadIdx.x; it < 32 * 128; it += 256) {
    int row = it >> 7;
    int ck  = (it & 127) * 8;
    int tok = assignTok[mBase + row];
    *(v8u*)(atile + row * LDS_STRIDE + ck) = *(const v8u*)(xbf + (size_t)tok * DIM + ck);
  }
  __syncthreads();

  GEMM_KLOOP(atile, w1bf)

  int rbase = (lane & 16) ? 8 : 0;
  int col = lane & 15;
  auto store_tile = [&](v8f c, int mi, int ni) {
    int ncol = nBase + ni * 16 + col;
    float bv = bias1[e * DIM + ncol];
    #pragma unroll
    for (int r = 0; r < 8; ++r) {
      float v = c[r] + bv;
      v = 0.5f * v * (1.0f + erff(v * 0.70710678118654752f));   // exact GELU
      hbuf[(size_t)(mBase + mi * 16 + rbase + r) * DIM + ncol] = f2bf(v);
    }
  };
  store_tile(c00, 0, 0);
  store_tile(c01, 0, 1);
  store_tile(c10, 1, 0);
  store_tile(c11, 1, 1);
}

// ---------------- GEMM2: Ya = H @ W2 + b2, f32 out ----------------
__global__ void __launch_bounds__(256)
gemm2_kernel(const u16t* __restrict__ hbuf, const u16t* __restrict__ w2bf,
             const float* __restrict__ bias2, const int* __restrict__ meta,
             float* __restrict__ yabuf) {
  __shared__ u16t atile[32 * LDS_STRIDE];
  int e = blockIdx.z;
  int pc = meta[8 + e];
  int mt = blockIdx.y * 32;
  if (mt >= pc) return;
  int mBase = meta[16 + e] + mt;
  int wave = threadIdx.x >> 5;
  int lane = threadIdx.x & 31;
  int nBase = blockIdx.x * 256 + wave * 32;

  // cooperative fill of A tile from H (no gather)
  for (int it = threadIdx.x; it < 32 * 128; it += 256) {
    int row = it >> 7;
    int ck  = (it & 127) * 8;
    *(v8u*)(atile + row * LDS_STRIDE + ck) =
        *(const v8u*)(hbuf + (size_t)(mBase + row) * DIM + ck);
  }
  __syncthreads();

  GEMM_KLOOP(atile, w2bf)

  int rbase = (lane & 16) ? 8 : 0;
  int col = lane & 15;
  auto store_tile = [&](v8f c, int mi, int ni) {
    int ncol = nBase + ni * 16 + col;
    float bv = bias2[e * DIM + ncol];
    #pragma unroll
    for (int r = 0; r < 8; ++r)
      yabuf[(size_t)(mBase + mi * 16 + rbase + r) * DIM + ncol] = c[r] + bv;
  };
  store_tile(c00, 0, 0);
  store_tile(c01, 0, 1);
  store_tile(c10, 1, 0);
  store_tile(c11, 1, 1);
}

// ---------------- combine: y = g0*Ya[p0] + g1*Ya[p1] ----------------
__global__ void combine_kernel(const float* __restrict__ yabuf, const float* __restrict__ topGate,
                               const int* __restrict__ slotPos, float* __restrict__ yout) {
  int i = blockIdx.x * blockDim.x + threadIdx.x;   // N_TOK*DIM threads
  int n = i >> 10;
  int c = i & 1023;
  int p0 = slotPos[n * 2], p1 = slotPos[n * 2 + 1];
  yout[i] = topGate[n * 2] * yabuf[(size_t)p0 * DIM + c]
          + topGate[n * 2 + 1] * yabuf[(size_t)p1 * DIM + c];
}

// ---------------- aux loss (deterministic fixed-order reduction) ----------------
__global__ void loss_kernel(const int* __restrict__ topIdx, const float* __restrict__ topGate,
                            float* __restrict__ lossOut) {
  __shared__ float simp[256 * NEXP];
  __shared__ float sld[256 * NEXP];
  int tid = threadIdx.x;
  float imp[NEXP], ld[NEXP];
  #pragma unroll
  for (int e = 0; e < NEXP; ++e) { imp[e] = 0.0f; ld[e] = 0.0f; }
  for (int n = tid; n < N_TOK; n += 256) {
    #pragma unroll
    for (int s = 0; s < 2; ++s) {
      int e = topIdx[n * 2 + s];
      imp[e] += topGate[n * 2 + s];
      ld[e] += 1.0f;
    }
  }
  #pragma unroll
  for (int e = 0; e < NEXP; ++e) { simp[tid * NEXP + e] = imp[e]; sld[tid * NEXP + e] = ld[e]; }
  __syncthreads();
  for (int s = 128; s > 0; s >>= 1) {
    if (tid < s) {
      #pragma unroll
      for (int e = 0; e < NEXP; ++e) {
        simp[tid * NEXP + e] += simp[(tid + s) * NEXP + e];
        sld[tid * NEXP + e]  += sld[(tid + s) * NEXP + e];
      }
    }
    __syncthreads();
  }
  if (tid == 0) {
    float cv[2];
    for (int a = 0; a < 2; ++a) {
      const float* v = (a == 0) ? simp : sld;
      float sum = 0.0f;
      for (int e = 0; e < NEXP; ++e) sum += v[e];
      float mean = sum / NEXP;
      float var = 0.0f;
      for (int e = 0; e < NEXP; ++e) { float d = v[e] - mean; var += d * d; }
      var /= (NEXP - 1);                       // ddof=1
      cv[a] = var / (mean * mean + 1e-10f);
    }
    lossOut[0] = 0.01f * (cv[0] + cv[1]);
  }
}

extern "C" void kernel_launch(void* const* d_in, const int* in_sizes, int n_in,
                              void* d_out, int out_size, void* d_ws, size_t ws_size,
                              hipStream_t stream) {
  const float* x  = (const float*)d_in[0];
  const float* wg = (const float*)d_in[1];
  const float* W1 = (const float*)d_in[2];
  const float* b1 = (const float*)d_in[3];
  const float* W2 = (const float*)d_in[4];
  const float* b2 = (const float*)d_in[5];
  float* yout = (float*)d_out;

  char* ws = (char*)d_ws;
  size_t o = 0;
  auto take = [&](size_t bytes) -> char* {
    char* p = ws + o;
    o = (o + bytes + 255) & ~(size_t)255;
    return p;
  };
  u16t*  xbf        = (u16t*) take((size_t)N_TOK * DIM * 2);
  u16t*  w1bf       = (u16t*) take((size_t)NEXP * DIM * DIM * 2);
  u16t*  w2bf       = (u16t*) take((size_t)NEXP * DIM * DIM * 2);
  u16t*  hbuf       = (u16t*) take((size_t)CAP * DIM * 2);
  float* yabuf      = (float*)take((size_t)CAP * DIM * 4);
  int*   topIdx     = (int*)  take((size_t)N_TOK * 2 * 4);
  float* topGate    = (float*)take((size_t)N_TOK * 2 * 4);
  int*   assignTok  = (int*)  take((size_t)CAP * 4);
  float* assignGate = (float*)take((size_t)CAP * 4);
  int*   slotPos    = (int*)  take((size_t)N_TOK * 2 * 4);
  int*   meta       = (int*)  take(32 * 4);   // cnt[8], cntPad[8], offs[8], cursor[8]

  init_kernel<<<(CAP + 255) / 256, 256, 0, stream>>>(assignTok, assignGate, meta);
  conv_x_kernel<<<(N_TOK * DIM) / 256, 256, 0, stream>>>(x, xbf);
  prep_w_kernel<<<(NEXP * DIM * DIM) / 256, 256, 0, stream>>>(W1, w1bf);
  prep_w_kernel<<<(NEXP * DIM * DIM) / 256, 256, 0, stream>>>(W2, w2bf);
  gate_kernel<<<N_TOK / 8, 256, 0, stream>>>(x, wg, topIdx, topGate, meta);
  offsets_kernel<<<1, 32, 0, stream>>>(meta);
  scatter_kernel<<<N_TOK / 256, 256, 0, stream>>>(topIdx, topGate, meta,
                                                  assignTok, assignGate, slotPos);
  gemm1_kernel<<<dim3(4, 128, NEXP), 256, 0, stream>>>(xbf, w1bf, b1, assignTok, meta, hbuf);
  gemm2_kernel<<<dim3(4, 128, NEXP), 256, 0, stream>>>(hbuf, w2bf, b2, meta, yabuf);
  combine_kernel<<<(N_TOK * DIM) / 256, 256, 0, stream>>>(yabuf, topGate, slotPos, yout);
  loss_kernel<<<1, 256, 0, stream>>>(topIdx, topGate, yout + (size_t)N_TOK * DIM);
}
